// NER_17824114278517
// MI455X (gfx1250) — compile-verified, matrix operand order checked by reading, not run on hardware
//
#include <hip/hip_runtime.h>
#include <hip/hip_bf16.h>
#include <stdint.h>

#define T_LEN   4096
#define EMBED   1024
#define HID     1024
#define G4      4096      // 4*HID gate rows
#define C_TAGS  16
#define START_TAG 14
#define STOP_TAG  15
#define NEG_INF  (-10000.0f)
#define CLS_HID  256
#define NWG_REC  256
#define HPW      4        // hidden units per workgroup in recurrence

typedef __bf16 bf16;
typedef __attribute__((ext_vector_type(16))) __bf16 v16bf;
typedef __attribute__((ext_vector_type(8)))  __bf16 v8bf;
typedef __attribute__((ext_vector_type(8)))  float  v8f;

__device__ __forceinline__ unsigned short f2bf(float f) {
  unsigned int u = __float_as_uint(f);
  unsigned int lsb = (u >> 16) & 1u;
  return (unsigned short)((u + 0x7FFFu + lsb) >> 16);  // round-to-nearest-even
}

// ---------------------------------------------------------------------------
// LayerNorm (optionally via gather index) -> bf16 output row. One block per row.
// ---------------------------------------------------------------------------
__global__ __launch_bounds__(256) void ln_to_bf16_k(
    const float* __restrict__ in, const int* __restrict__ idx,
    const float* __restrict__ g, const float* __restrict__ b,
    unsigned short* __restrict__ out, int D)
{
  int t = blockIdx.x;
  const float* row = in + (size_t)(idx ? idx[t] : t) * D;
  float s = 0.f, s2 = 0.f;
  for (int i = threadIdx.x; i < D; i += 256) { float v = row[i]; s += v; s2 += v * v; }
#pragma unroll
  for (int m = 16; m >= 1; m >>= 1) { s += __shfl_xor(s, m, 32); s2 += __shfl_xor(s2, m, 32); }
  __shared__ float red[16];
  __shared__ float stats[2];
  int wave = threadIdx.x >> 5;
  if ((threadIdx.x & 31) == 0) { red[wave] = s; red[8 + wave] = s2; }
  __syncthreads();
  if (threadIdx.x == 0) {
    float S = 0.f, S2 = 0.f;
    for (int w = 0; w < 8; ++w) { S += red[w]; S2 += red[8 + w]; }
    float mean = S / (float)D;
    float var  = S2 / (float)D - mean * mean;
    stats[0] = mean; stats[1] = rsqrtf(var + 1e-5f);
  }
  __syncthreads();
  float mean = stats[0], rstd = stats[1];
  for (int i = threadIdx.x; i < D; i += 256) {
    float y = (row[i] - mean) * rstd * g[i] + b[i];
    out[(size_t)t * D + i] = f2bf(y);
  }
}

// ---------------------------------------------------------------------------
// f32 -> bf16 elementwise
// ---------------------------------------------------------------------------
__global__ __launch_bounds__(256) void f32_to_bf16_k(
    const float* __restrict__ in, unsigned short* __restrict__ out, int n)
{
  int i = blockIdx.x * 256 + threadIdx.x;
  if (i < n) out[i] = f2bf(in[i]);
}

// ---------------------------------------------------------------------------
// Load one 16x32 bf16 WMMA operand fragment (two contiguous 16B chunks).
// ISA layout: lane<16 holds row=lane, K chunks {cs..cs+7, cs+16..cs+23} with
// cs = 0 (lane<16) or 8 (lane>=16).
// ---------------------------------------------------------------------------
__device__ __forceinline__ v16bf load_frag(const bf16* __restrict__ p) {
  v8bf lo = *reinterpret_cast<const v8bf*>(p);
  v8bf hi = *reinterpret_cast<const v8bf*>(p + 16);
  v16bf r;
#pragma unroll
  for (int i = 0; i < 8; ++i) { r[i] = lo[i]; r[i + 8] = hi[i]; }
  return r;
}

// ---------------------------------------------------------------------------
// Register-blocked WMMA GEMM: C[M,N] = A[M,K](bf16) * B[N,K](bf16)^T + bias.
// Each wave owns a 32x64 output strip = 2x4 tiles of 16x16: per K=32 step,
// 12 b128 loads feed 8 v_wmma_f32_16x16x32_bf16 (A reused 4x, B reused 2x).
// ---------------------------------------------------------------------------
__global__ __launch_bounds__(256) void wmma_gemm_bf16_k(
    const unsigned short* __restrict__ Au, const unsigned short* __restrict__ Bu,
    const float* __restrict__ bias, float* __restrict__ Cout,
    int M, int N, int K, int leaky)
{
  const bf16* A = reinterpret_cast<const bf16*>(Au);
  const bf16* B = reinterpret_cast<const bf16*>(Bu);
  int wave = threadIdx.x >> 5;
  int lane = threadIdx.x & 31;
  int stripsN = N >> 6;                     // 64-wide strips
  int strip = blockIdx.x * 8 + wave;        // wave-uniform: EXEC stays all-ones
  if (strip >= (M >> 5) * stripsN) return;
  int m0 = (strip / stripsN) << 5;
  int n0 = (strip % stripsN) << 6;
  int l15 = lane & 15;
  int hi  = lane >> 4;
  int cs  = hi << 3;                        // K sub-chunk offset 0 or 8
  const bf16* ap[2];
  const bf16* bp[4];
#pragma unroll
  for (int mt = 0; mt < 2; ++mt) ap[mt] = A + (size_t)(m0 + mt * 16 + l15) * K + cs;
#pragma unroll
  for (int nt = 0; nt < 4; ++nt) bp[nt] = B + (size_t)(n0 + nt * 16 + l15) * K + cs;

  v8f acc[2][4] = {};
  for (int k0 = 0; k0 < K; k0 += 32) {
    v16bf af[2], bf[4];
#pragma unroll
    for (int mt = 0; mt < 2; ++mt) af[mt] = load_frag(ap[mt] + k0);
#pragma unroll
    for (int nt = 0; nt < 4; ++nt) bf[nt] = load_frag(bp[nt] + k0);
#pragma unroll
    for (int mt = 0; mt < 2; ++mt)
#pragma unroll
      for (int nt = 0; nt < 4; ++nt)
        acc[mt][nt] = __builtin_amdgcn_wmma_f32_16x16x32_bf16(
            false, af[mt], false, bf[nt], (short)0, acc[mt][nt], false, false);
  }
#pragma unroll
  for (int mt = 0; mt < 2; ++mt) {
    int r0 = m0 + mt * 16 + (hi << 3);      // C/D layout: M = r + 8*(lane>=16)
#pragma unroll
    for (int nt = 0; nt < 4; ++nt) {
      int col = n0 + nt * 16 + l15;
      float bb = bias ? bias[col] : 0.0f;
#pragma unroll
      for (int r = 0; r < 8; ++r) {
        float v = acc[mt][nt][r] + bb;
        if (leaky) v = (v > 0.0f) ? v : 0.01f * v;
        Cout[(size_t)(r0 + r) * N + col] = v;
      }
    }
  }
}

// ---------------------------------------------------------------------------
// Grid-wide sense-reversal barrier (persistent recurrence kernel).
// ---------------------------------------------------------------------------
__device__ __forceinline__ void grid_barrier(int* counter, int* generation, int nwg)
{
  __syncthreads();
  if (threadIdx.x == 0) {
    __threadfence();
    int gen  = __hip_atomic_load(generation, __ATOMIC_ACQUIRE, __HIP_MEMORY_SCOPE_AGENT);
    int prev = __hip_atomic_fetch_add(counter, 1, __ATOMIC_ACQ_REL, __HIP_MEMORY_SCOPE_AGENT);
    if (prev == nwg - 1) {
      __hip_atomic_store(counter, 0, __ATOMIC_RELAXED, __HIP_MEMORY_SCOPE_AGENT);
      __hip_atomic_store(generation, gen + 1, __ATOMIC_RELEASE, __HIP_MEMORY_SCOPE_AGENT);
    } else {
      while (__hip_atomic_load(generation, __ATOMIC_ACQUIRE, __HIP_MEMORY_SCOPE_AGENT) == gen)
        __builtin_amdgcn_s_sleep(2);
    }
  }
  __syncthreads();
}

// ---------------------------------------------------------------------------
// LSTM recurrence: persistent grid, 256 WGs x 256 threads.
// WG w owns hidden units [4w,4w+4) => 16 gate rows of Whh (i,f,g,o blocks).
// Whh streams from the 192MB L2 each step; the 4KB h vector is staged into
// LDS with the CDNA5 async memory->LDS DMA (global_load_async_to_lds_b128,
// ASYNCcnt-tracked), then 16-lane float4 dots reduced via shfl_xor.
// ---------------------------------------------------------------------------
__global__ __launch_bounds__(256) void lstm_rec_k(
    const float* __restrict__ xw,     // T x 4096, includes input bias
    const float* __restrict__ Whh,    // 4096 x 1024
    float* __restrict__ outf,         // T x outStride
    unsigned short* outbf,            // optional bf16 copy (T x 2048), may be null
    float* __restrict__ hbuf,         // ping-pong 2 x 1024
    int* __restrict__ bar,            // {counter, generation}
    int outStride, int outCol, int reverse)
{
  __shared__ float h_s[HID];
  __shared__ float gsum[16];
  __shared__ float c_s[HPW];
  int w = blockIdx.x;
  int tid = threadIdx.x;
  if (tid < HPW) {
    c_s[tid] = 0.f;
    hbuf[w * HPW + tid] = 0.f;
    hbuf[HID + w * HPW + tid] = 0.f;
  }
  int r  = tid >> 4;                       // local gate row 0..15
  int th = tid & 15;                       // 16 lanes per row
  int grow = (r >> 2) * HID + w * HPW + (r & 3);  // global gate row
  const float* wr = Whh + (size_t)grow * HID;
  // LDS byte offset of h_s: low 32 bits of the flat shared-aperture address
  unsigned hs_base = (unsigned)(uintptr_t)(&h_s[0]);
  grid_barrier(bar, bar + 1, NWG_REC);

  for (int t = 0; t < T_LEN; ++t) {
    int tt = reverse ? (T_LEN - 1 - t) : t;
    // ---- async DMA: 256 threads x 16B = full h vector, memory -> LDS ----
    {
      unsigned ldsa = hs_base + (unsigned)(tid << 4);
      const float* gp = hbuf + (size_t)(t & 1) * HID + (tid << 2);
      asm volatile("global_load_async_to_lds_b128 %0, %1, off"
                   :: "v"(ldsa), "v"(gp) : "memory");
#if __has_builtin(__builtin_amdgcn_s_wait_asynccnt)
      __builtin_amdgcn_s_wait_asynccnt(0);
#else
      asm volatile("s_wait_asynccnt 0x0" ::: "memory");
#endif
    }
    __syncthreads();
    float p = 0.f;
#pragma unroll 4
    for (int kk = 0; kk < 16; ++kk) {      // coalesced float4 sweep of the row
      int k = kk * 64 + th * 4;
      float4 wv = *reinterpret_cast<const float4*>(wr + k);
      p += wv.x * h_s[k] + wv.y * h_s[k + 1] + wv.z * h_s[k + 2] + wv.w * h_s[k + 3];
    }
#pragma unroll
    for (int m = 8; m >= 1; m >>= 1) p += __shfl_xor(p, m, 32);
    if (th == 0) gsum[r] = p;
    __syncthreads();
    if (tid < HPW) {
      int idx = w * HPW + tid;
      const float* xt = xw + (size_t)tt * G4;
      float gi = xt[idx]           + gsum[tid];
      float gf = xt[HID + idx]     + gsum[4 + tid];
      float gg = xt[2 * HID + idx] + gsum[8 + tid];
      float go = xt[3 * HID + idx] + gsum[12 + tid];
      gi = 1.f / (1.f + __expf(-gi));
      gf = 1.f / (1.f + __expf(-gf));
      go = 1.f / (1.f + __expf(-go));
      gg = tanhf(gg);
      float c = gf * c_s[tid] + gi * gg;
      c_s[tid] = c;
      float h = go * tanhf(c);
      hbuf[((t + 1) & 1) * HID + idx] = h;
      outf[(size_t)tt * outStride + outCol + idx] = h;
      if (outbf) outbf[(size_t)tt * 2048 + outCol + idx] = f2bf(h);
      int tn = reverse ? (T_LEN - 2 - t) : (t + 1);
      if (tn >= 0 && tn < T_LEN)
        __builtin_prefetch(xw + (size_t)tn * G4 + idx, 0, 1);  // global_prefetch_b8
    }
    grid_barrier(bar, bar + 1, NWG_REC);
  }
}

// ---------------------------------------------------------------------------
// Tiny 256->16 FC: one thread per (t, tag)
// ---------------------------------------------------------------------------
__global__ __launch_bounds__(256) void small_fc_k(
    const float* __restrict__ h1, const float* __restrict__ w2,
    const float* __restrict__ b2, float* __restrict__ feats)
{
  int i = blockIdx.x * 256 + threadIdx.x;
  if (i >= T_LEN * C_TAGS) return;
  int t = i >> 4, c = i & 15;
  const float* hr = h1 + (size_t)t * CLS_HID;
  const float* wr = w2 + (size_t)c * CLS_HID;
  float s = b2[c];
  for (int k = 0; k < CLS_HID; ++k) s += hr[k] * wr[k];
  feats[i] = s;
}

// ---------------------------------------------------------------------------
// Viterbi: single wave; lane = next-state; byte backpointers; serial backtrack.
// ---------------------------------------------------------------------------
__global__ void viterbi_k(const float* __restrict__ feats,
                          const float* __restrict__ trans,
                          unsigned char* __restrict__ bptr,
                          float* __restrict__ out, int Tlen)
{
  __shared__ float tr[C_TAGS][C_TAGS];
  __shared__ float fv[C_TAGS];
  int tid = threadIdx.x;
  for (int i = tid; i < C_TAGS * C_TAGS; i += 32) tr[i >> 4][i & 15] = trans[i];
  if (tid < C_TAGS) fv[tid] = (tid == START_TAG) ? 0.f : NEG_INF;
  __syncthreads();
  for (int t = 0; t < Tlen; ++t) {
    float best = 0.f; int bi = 0;
    if (tid < C_TAGS) {
      best = fv[0] + tr[tid][0];
      for (int pv = 1; pv < C_TAGS; ++pv) {
        float sc = fv[pv] + tr[tid][pv];
        if (sc > best) { best = sc; bi = pv; }        // first-max like jnp.argmax
      }
    }
    __syncthreads();
    if (tid < C_TAGS) {
      fv[tid] = best + feats[(size_t)t * C_TAGS + tid];
      bptr[(size_t)t * C_TAGS + tid] = (unsigned char)bi;
    }
    __syncthreads();
  }
  if (tid == 0) {
    float bestS = fv[0] + tr[STOP_TAG][0]; int bt = 0;
    for (int c = 1; c < C_TAGS; ++c) {
      float s = fv[c] + tr[STOP_TAG][c];
      if (s > bestS) { bestS = s; bt = c; }
    }
    out[0] = bestS;
    int tag = bt;
    for (int t = Tlen - 1; t >= 0; --t) {
      out[1 + t] = (float)tag;
      tag = bptr[(size_t)t * C_TAGS + tag];
    }
  }
}

// ---------------------------------------------------------------------------
extern "C" void kernel_launch(void* const* d_in, const int* in_sizes, int n_in,
                              void* d_out, int out_size, void* d_ws, size_t ws_size,
                              hipStream_t stream)
{
  (void)in_sizes; (void)n_in; (void)out_size; (void)ws_size;
  const int*   x     = (const int*)  d_in[0];
  const float* table = (const float*)d_in[1];
  const float* ln1_g = (const float*)d_in[2];
  const float* ln1_b = (const float*)d_in[3];
  // lstm flatten order: (L0 fwd Wih,Whh,b), (L0 bwd ...), (L1 fwd ...), (L1 bwd ...)
  const float* Wih[4] = {(const float*)d_in[4],  (const float*)d_in[7],
                         (const float*)d_in[10], (const float*)d_in[13]};
  const float* Whh[4] = {(const float*)d_in[5],  (const float*)d_in[8],
                         (const float*)d_in[11], (const float*)d_in[14]};
  const float* bg [4] = {(const float*)d_in[6],  (const float*)d_in[9],
                         (const float*)d_in[12], (const float*)d_in[15]};
  const float* ln2_g = (const float*)d_in[16];
  const float* ln2_b = (const float*)d_in[17];
  const float* w1    = (const float*)d_in[18];
  const float* b1    = (const float*)d_in[19];
  const float* w2    = (const float*)d_in[20];
  const float* b2    = (const float*)d_in[21];
  const float* trans = (const float*)d_in[22];

  char* p = (char*)d_ws;
  auto take = [&](size_t bytes) { char* r = p; p += (bytes + 255) & ~(size_t)255; return r; };
  unsigned short* ebf    = (unsigned short*)take((size_t)T_LEN * EMBED * 2);
  unsigned short* wbf    = (unsigned short*)take((size_t)G4 * 2048 * 2);   // weight scratch (max Wih)
  float*          xw     = (float*)         take((size_t)T_LEN * G4 * 4);  // reused per direction
  float*          out1f  = (float*)         take((size_t)T_LEN * 2048 * 4);
  unsigned short* out1bf = (unsigned short*)take((size_t)T_LEN * 2048 * 2);
  float*          out2f  = (float*)         take((size_t)T_LEN * 2048 * 4);
  unsigned short* hnbf   = (unsigned short*)take((size_t)T_LEN * 2048 * 2);
  float*          h1     = (float*)         take((size_t)T_LEN * CLS_HID * 4);
  float*          feats  = (float*)         take((size_t)T_LEN * C_TAGS * 4);
  float*          hbuf   = (float*)         take((size_t)2 * HID * 4);
  int*            bar    = (int*)           take(256);
  unsigned char*  bptr   = (unsigned char*) take((size_t)T_LEN * C_TAGS);

  hipMemsetAsync(bar, 0, 256, stream);

  // Embedding gather + LN1 -> bf16 activations
  ln_to_bf16_k<<<T_LEN, 256, 0, stream>>>(table, x, ln1_g, ln1_b, ebf, EMBED);

  // 32x64 strip per wave, 8 waves per block
  const int gemmBlocksBig = ((T_LEN / 32) * (G4 / 64)) / 8;   // 1024 blocks
  const int gemmBlocksCls = ((T_LEN / 32) * (CLS_HID / 64)) / 8;

  // Layer 1 forward
  f32_to_bf16_k<<<(G4 * EMBED + 255) / 256, 256, 0, stream>>>(Wih[0], wbf, G4 * EMBED);
  wmma_gemm_bf16_k<<<gemmBlocksBig, 256, 0, stream>>>(ebf, wbf, bg[0], xw, T_LEN, G4, EMBED, 0);
  lstm_rec_k<<<NWG_REC, 256, 0, stream>>>(xw, Whh[0], out1f, out1bf, hbuf, bar, 2048, 0, 0);
  // Layer 1 backward
  f32_to_bf16_k<<<(G4 * EMBED + 255) / 256, 256, 0, stream>>>(Wih[1], wbf, G4 * EMBED);
  wmma_gemm_bf16_k<<<gemmBlocksBig, 256, 0, stream>>>(ebf, wbf, bg[1], xw, T_LEN, G4, EMBED, 0);
  lstm_rec_k<<<NWG_REC, 256, 0, stream>>>(xw, Whh[1], out1f, out1bf, hbuf, bar, 2048, HID, 1);
  // Layer 2 forward
  f32_to_bf16_k<<<(G4 * 2048 + 255) / 256, 256, 0, stream>>>(Wih[2], wbf, G4 * 2048);
  wmma_gemm_bf16_k<<<gemmBlocksBig, 256, 0, stream>>>(out1bf, wbf, bg[2], xw, T_LEN, G4, 2048, 0);
  lstm_rec_k<<<NWG_REC, 256, 0, stream>>>(xw, Whh[2], out2f, nullptr, hbuf, bar, 2048, 0, 0);
  // Layer 2 backward
  f32_to_bf16_k<<<(G4 * 2048 + 255) / 256, 256, 0, stream>>>(Wih[3], wbf, G4 * 2048);
  wmma_gemm_bf16_k<<<gemmBlocksBig, 256, 0, stream>>>(out1bf, wbf, bg[3], xw, T_LEN, G4, 2048, 0);
  lstm_rec_k<<<NWG_REC, 256, 0, stream>>>(xw, Whh[3], out2f, nullptr, hbuf, bar, 2048, HID, 1);

  // LN2 -> bf16, classifier GEMM (leaky relu fused), tiny FC, Viterbi
  ln_to_bf16_k<<<T_LEN, 256, 0, stream>>>(out2f, nullptr, ln2_g, ln2_b, hnbf, 2048);
  f32_to_bf16_k<<<(CLS_HID * 2048 + 255) / 256, 256, 0, stream>>>(w1, wbf, CLS_HID * 2048);
  wmma_gemm_bf16_k<<<gemmBlocksCls, 256, 0, stream>>>(hnbf, wbf, b1, h1, T_LEN, CLS_HID, 2048, 1);
  small_fc_k<<<(T_LEN * C_TAGS + 255) / 256, 256, 0, stream>>>(h1, w2, b2, feats);
  viterbi_k<<<1, 32, 0, stream>>>(feats, trans, bptr, (float*)d_out, T_LEN);
}
